// GatedAxialAttentionBlock_45913200394524
// MI455X (gfx1250) — matrix operand order repaired and checked by reading, not compiled, not run on hardware
//
#include <hip/hip_runtime.h>
#include <hip/hip_bf16.h>

typedef _Float16 h16;
typedef __attribute__((ext_vector_type(16))) _Float16 v16h;
typedef __attribute__((ext_vector_type(8)))  float    v8f;

#define EPSV 1e-5f

// ---- workspace layout (bytes) ----
// y   f16 [b][h][o][w]   : 4*128*128*128*2 = 16,777,216   at 0
// yn  f16 [b][w][o][h]   : 16,777,216                     at 16,777,216
// attv  f32 [n][h][c][p] : 16,777,216                     at 33,554,432
// attvp f32 [n][h][c][p] : 16,777,216                     at 50,331,648
// stats f32[...]                                           at 67,108,864
#define ST_BNA   0    // 128
#define ST_BNB   128  // 128
#define ST_S24   256  // 24 sums
#define ST_SS24  280  // 24 sumsq
#define ST_CA    304  // 24 BN-att scales
#define ST_CST   328  // 8 combined biases
#define ST_SV    336  // 16 sums
#define ST_SSV   352  // 16 sumsq
#define ST_AV    368  // 16
#define ST_BV    384  // 16

__device__ inline float wsum32(float v) {
#pragma unroll
  for (int m = 16; m; m >>= 1) v += __shfl_xor(v, m, 32);
  return v;
}
__device__ inline float wmax32(float v) {
#pragma unroll
  for (int m = 16; m; m >>= 1) v = fmaxf(v, __shfl_xor(v, m, 32));
  return v;
}

// -------------------- pass 0: zero stats --------------------
__global__ void k_zero(float* st) {
  int t = threadIdx.x;
  if (t < 48) st[ST_S24 + t] = 0.f;   // S24 + SS24
  if (t < 32) st[ST_SV + t] = 0.f;    // SV + SSV
}

// -------------------- pass 1: y = relu(W @ x), per (b,h) block --------------------
// out: y[b][h][o][w] f16
__global__ __launch_bounds__(256) void k_gemm_qkv(const float* __restrict__ x,
                                                  const float* __restrict__ wq,
                                                  h16* __restrict__ y) {
  __shared__ h16 Ws[128][68];   // [o][c], padded
  __shared__ h16 XsT[128][68];  // [w][c], padded
  const int b = blockIdx.x >> 7, h = blockIdx.x & 127;
  const int tid = threadIdx.x;
  for (int j = tid; j < 128 * 64; j += 256)
    Ws[j >> 6][j & 63] = (h16)wq[j];
  for (int j = tid; j < 64 * 128; j += 256) {
    int c = j >> 7, w = j & 127;
    XsT[w][c] = (h16)x[(((size_t)b * 64 + c) * 128 + h) * 128 + w];
  }
  __syncthreads();

  const int wv = tid >> 5, l = tid & 31;
  const int lm = l & 15, hi = l >> 4;
  const int M = wv * 16 + lm;

  v8f acc[8];
#pragma unroll
  for (int t = 0; t < 8; ++t)
#pragma unroll
    for (int r = 0; r < 8; ++r) acc[t][r] = 0.f;

#pragma unroll
  for (int ks = 0; ks < 2; ++ks) {
    const int k0 = ks * 32;
    v16h a;
    {
      const h16* row = &Ws[M][0];
      const int kb = k0 + hi * 16;
#pragma unroll
      for (int e = 0; e < 8; ++e) { a[e] = row[kb + e]; a[8 + e] = row[kb + 8 + e]; }
    }
#pragma unroll
    for (int t = 0; t < 8; ++t) {
      const int w0 = t * 16;
      v16h bm;
      const h16* col = &XsT[w0 + lm][0];
      const int kb = k0 + hi * 16;
#pragma unroll
      for (int e = 0; e < 16; ++e) bm[e] = col[kb + e];
      acc[t] = __builtin_amdgcn_wmma_f32_16x16x32_f16(false, a, false, bm,
                                                      (short)0, acc[t], false, false);
    }
  }
  // relu + store f16
  const size_t base = (size_t)blockIdx.x * 128 * 128;  // (b*128+h)*128*128
#pragma unroll
  for (int t = 0; t < 8; ++t) {
    const int w = t * 16 + lm;
#pragma unroll
    for (int r = 0; r < 8; ++r) {
      const int o = wv * 16 + hi * 8 + r;
      float v = acc[t][r];
      v = v > 0.f ? v : 0.f;
      y[base + (size_t)o * 128 + w] = (h16)v;
    }
  }
}

// -------------------- pass 2: BN1 stats, one block per channel o --------------------
__global__ __launch_bounds__(256) void k_stats1(const h16* __restrict__ y,
                                                const float* __restrict__ g1,
                                                const float* __restrict__ b1,
                                                float* __restrict__ st) {
  const int o = blockIdx.x, tid = threadIdx.x;
  float s = 0.f, ss = 0.f;
  for (int j = tid; j < 512 * 128; j += 256) {
    const int bh = j >> 7, w = j & 127;
    const float v = (float)y[((size_t)bh * 128 + o) * 128 + w];
    s += v; ss += v * v;
  }
  __shared__ float rs[256], rss[256];
  rs[tid] = s; rss[tid] = ss;
  __syncthreads();
  for (int d = 128; d > 0; d >>= 1) {
    if (tid < d) { rs[tid] += rs[tid + d]; rss[tid] += rss[tid + d]; }
    __syncthreads();
  }
  if (tid == 0) {
    const float cnt = 65536.f;
    const float mean = rs[0] / cnt;
    const float var = rss[0] / cnt - mean * mean;
    const float rstd = rsqrtf(var + EPSV);
    const float g = g1[o];
    st[ST_BNA + o] = g * rstd;
    st[ST_BNB + o] = b1[o] - g * mean * rstd;
  }
}

// -------------------- pass 3: BN apply + h<->w transpose --------------------
// in: y[b][h][o][w], out: yn[b][w][o][h], block per (b,o)
__global__ __launch_bounds__(256) void k_bn_transpose(const h16* __restrict__ y,
                                                      const float* __restrict__ st,
                                                      h16* __restrict__ yn) {
  const int b = blockIdx.x >> 7, o = blockIdx.x & 127;
  const int tid = threadIdx.x;
  __shared__ h16 ts[128][132];
  const float a = st[ST_BNA + o], bb = st[ST_BNB + o];
  for (int j = tid; j < 16384; j += 256) {
    const int hh = j >> 7, w = j & 127;
    const float v = (float)y[((size_t)(b * 128 + hh) * 128 + o) * 128 + w];
    ts[hh][w] = (h16)fmaf(a, v, bb);
  }
  __syncthreads();
  for (int j = tid; j < 16384; j += 256) {
    const int w = j >> 7, hh = j & 127;
    yn[((size_t)(b * 128 + w) * 128 + o) * 128 + hh] = ts[hh][w];
  }
}

// -------------------- pass 4: att stats (qk/qr/kr), block per (n,h) --------------------
__global__ __launch_bounds__(256) void k_att_stats(const h16* __restrict__ yn,
                                                   const float* __restrict__ pr,
                                                   const float* __restrict__ gqr_p,
                                                   const float* __restrict__ gkr_p,
                                                   float* __restrict__ st) {
  const int n = blockIdx.x >> 3, h = blockIdx.x & 7;
  const int tid = threadIdx.x;
  __shared__ float qs[8][128];
  __shared__ float ps[8][255];
  __shared__ float acc6[6];
  if (tid < 6) acc6[tid] = 0.f;
  for (int j = tid; j < 8 * 128; j += 256) {
    const int c = j >> 7, i = j & 127;
    qs[c][i] = (float)yn[((size_t)n * 128 + h * 16 + c) * 128 + i];
  }
  for (int j = tid; j < 8 * 255; j += 256)
    ps[j / 255][j % 255] = pr[j];
  __syncthreads();
  const float gqr = gqr_p[0], gkr = gkr_p[0];
  const int p = tid >> 1, i0 = (tid & 1) * 64;
  const float qp0 = qs[0][p], qp1 = qs[1][p], qp2 = qs[2][p], qp3 = qs[3][p];
  const float kp0 = qs[4][p], kp1 = qs[5][p], kp2 = qs[6][p], kp3 = qs[7][p];
  float s0 = 0, s1 = 0, s2 = 0, t0 = 0, t1 = 0, t2 = 0;
  for (int ii = 0; ii < 64; ++ii) {
    const int i = i0 + ii;
    const int d = p - i + 127;
    const float qk = qp0 * qs[4][i] + qp1 * qs[5][i] + qp2 * qs[6][i] + qp3 * qs[7][i];
    const float qr = gqr * (qp0 * ps[0][d] + qp1 * ps[1][d] + qp2 * ps[2][d] + qp3 * ps[3][d]);
    const float kr = gkr * (kp0 * ps[4][d] + kp1 * ps[5][d] + kp2 * ps[6][d] + kp3 * ps[7][d]);
    s0 += qk; t0 += qk * qk;
    s1 += qr; t1 += qr * qr;
    s2 += kr; t2 += kr * kr;
  }
  s0 = wsum32(s0); s1 = wsum32(s1); s2 = wsum32(s2);
  t0 = wsum32(t0); t1 = wsum32(t1); t2 = wsum32(t2);
  if ((tid & 31) == 0) {
    atomicAdd(&acc6[0], s0); atomicAdd(&acc6[1], s1); atomicAdd(&acc6[2], s2);
    atomicAdd(&acc6[3], t0); atomicAdd(&acc6[4], t1); atomicAdd(&acc6[5], t2);
  }
  __syncthreads();
  if (tid == 0) {
    atomicAdd(&st[ST_S24 + h],       acc6[0]); atomicAdd(&st[ST_SS24 + h],       acc6[3]);
    atomicAdd(&st[ST_S24 + 8 + h],   acc6[1]); atomicAdd(&st[ST_SS24 + 8 + h],   acc6[4]);
    atomicAdd(&st[ST_S24 + 16 + h],  acc6[2]); atomicAdd(&st[ST_SS24 + 16 + h],  acc6[5]);
  }
}

// -------------------- pass 5: finalize att BN coefficients --------------------
__global__ void k_fin_att(const float* __restrict__ ga, const float* __restrict__ be,
                          float* __restrict__ st) {
  const int t = threadIdx.x;
  __shared__ float cb[24];
  if (t < 24) {
    const float cnt = 512.f * 128.f * 128.f;
    const float mean = st[ST_S24 + t] / cnt;
    const float var = st[ST_SS24 + t] / cnt - mean * mean;
    const float rstd = rsqrtf(var + EPSV);
    const float g = ga[t];
    st[ST_CA + t] = g * rstd;
    cb[t] = be[t] - g * mean * rstd;
  }
  __syncthreads();
  if (t < 8) st[ST_CST + t] = cb[t] + cb[8 + t] + cb[16 + t];
}

// -------------------- pass 6: att apply + softmax + att@v (WMMA) + att@v_pos ---------
#define ATT_STR 130
#define ATTH_STR 132
#define SM_ATT   0
#define SM_ATTH  (128 * ATT_STR * 4)
#define SM_PS    (SM_ATTH + 128 * ATTH_STR * 2)
#define SM_QS    (SM_PS + 16 * 255 * 4)
#define SM_VS    (SM_QS + 8 * 128 * 4)
#define SM_RED   (SM_VS + 16 * ATTH_STR * 2)
#define SM_BYTES (SM_RED + 16 * 4)

__global__ __launch_bounds__(256) void k_att_apply(const h16* __restrict__ yn,
                                                   const float* __restrict__ pr,
                                                   const float* __restrict__ gqr_p,
                                                   const float* __restrict__ gkr_p,
                                                   const float* __restrict__ gvp_p,
                                                   const float* __restrict__ gv_p,
                                                   float* __restrict__ st,
                                                   float* __restrict__ attv,
                                                   float* __restrict__ attvp) {
  extern __shared__ char smem[];
  float* att = (float*)(smem + SM_ATT);   // [128][ATT_STR]
  h16*  atth = (h16*)(smem + SM_ATTH);    // [128][ATTH_STR]
  float* ps  = (float*)(smem + SM_PS);    // [16][255]
  float* qs  = (float*)(smem + SM_QS);    // [8][128]
  h16*  vs   = (h16*)(smem + SM_VS);      // [16][ATTH_STR]
  float* red = (float*)(smem + SM_RED);

  const int n = blockIdx.x >> 3, h = blockIdx.x & 7;
  const int tid = threadIdx.x;
  const float gv = gv_p[0];
  if (tid < 4) red[tid] = 0.f;
  for (int j = tid; j < 8 * 128; j += 256) {
    const int c = j >> 7, i = j & 127;
    qs[c * 128 + i] = (float)yn[((size_t)n * 128 + h * 16 + c) * 128 + i];
  }
  for (int j = tid; j < 16 * ATTH_STR; j += 256) {
    const int c = j / ATTH_STR, i = j % ATTH_STR;
    h16 v = (h16)0.f;
    if (c < 8 && i < 128)
      v = (h16)((float)yn[((size_t)n * 128 + h * 16 + 8 + c) * 128 + i] * gv);
    vs[j] = v;
  }
  for (int j = tid; j < 16 * 255; j += 256) ps[j] = pr[j];
  __syncthreads();

  const float gqr = gqr_p[0], gkr = gkr_p[0], gvp = gvp_p[0];
  const float aqk = st[ST_CA + h], aqr = st[ST_CA + 8 + h], akr = st[ST_CA + 16 + h];
  const float cst = st[ST_CST + h];

  // combined pre-softmax attention
  {
    const int p = tid >> 1, i0 = (tid & 1) * 64;
    const float qp0 = qs[0 * 128 + p], qp1 = qs[1 * 128 + p], qp2 = qs[2 * 128 + p], qp3 = qs[3 * 128 + p];
    const float kp0 = qs[4 * 128 + p], kp1 = qs[5 * 128 + p], kp2 = qs[6 * 128 + p], kp3 = qs[7 * 128 + p];
    for (int ii = 0; ii < 64; ++ii) {
      const int i = i0 + ii;
      const int d = p - i + 127;
      const float qk = qp0 * qs[4 * 128 + i] + qp1 * qs[5 * 128 + i] + qp2 * qs[6 * 128 + i] + qp3 * qs[7 * 128 + i];
      const float qr = gqr * (qp0 * ps[0 * 255 + d] + qp1 * ps[1 * 255 + d] + qp2 * ps[2 * 255 + d] + qp3 * ps[3 * 255 + d]);
      const float kr = gkr * (kp0 * ps[4 * 255 + d] + kp1 * ps[5 * 255 + d] + kp2 * ps[6 * 255 + d] + kp3 * ps[7 * 255 + d]);
      att[p * ATT_STR + i] = fmaf(aqk, qk, fmaf(aqr, qr, fmaf(akr, kr, cst)));
    }
  }
  __syncthreads();

  // softmax per row: 8 waves x 16 rows
  const int wv = tid >> 5, l = tid & 31;
  for (int rr = 0; rr < 16; ++rr) {
    const int p = wv * 16 + rr;
    float v[4];
    float mx = -1e30f;
#pragma unroll
    for (int j = 0; j < 4; ++j) { v[j] = att[p * ATT_STR + 32 * j + l]; mx = fmaxf(mx, v[j]); }
    mx = wmax32(mx);
    float s = 0.f;
#pragma unroll
    for (int j = 0; j < 4; ++j) { v[j] = __expf(v[j] - mx); s += v[j]; }
    s = wsum32(s);
    const float inv = 1.f / s;
#pragma unroll
    for (int j = 0; j < 4; ++j) {
      const float e = v[j] * inv;
      att[p * ATT_STR + 32 * j + l] = e;
      atth[p * ATTH_STR + 32 * j + l] = (h16)e;
    }
  }
  __syncthreads();

  // att @ (v*gate_v)^T via WMMA: per wave a 16(p) x 16(c-padded) tile, K=128
  {
    const int lm = l & 15, hi = l >> 4;
    const int p0 = wv * 16;
    v8f acc;
#pragma unroll
    for (int r = 0; r < 8; ++r) acc[r] = 0.f;
#pragma unroll
    for (int ks = 0; ks < 4; ++ks) {
      const int k0 = ks * 32;
      v16h a;
      {
        const h16* row = &atth[(p0 + lm) * ATTH_STR];
        const int kb = k0 + hi * 16;
#pragma unroll
        for (int e = 0; e < 8; ++e) { a[e] = row[kb + e]; a[8 + e] = row[kb + 8 + e]; }
      }
      v16h bm;
      {
        const h16* col = &vs[lm * ATTH_STR];
        const int kb = k0 + hi * 16;
#pragma unroll
        for (int e = 0; e < 16; ++e) bm[e] = col[kb + e];
      }
      acc = __builtin_amdgcn_wmma_f32_16x16x32_f16(false, a, false, bm, (short)0, acc, false, false);
    }
    float sv = 0.f, ssv = 0.f;
    if (lm < 8) {
      const int c = lm;
      const int pb = p0 + hi * 8;
      float* dst = &attv[(((size_t)n * 8 + h) * 8 + c) * 128 + pb];
#pragma unroll
      for (int r = 0; r < 8; ++r) {
        const float v = acc[r];
        dst[r] = v;
        sv += v; ssv += v * v;
      }
    }
    sv = wsum32(sv); ssv = wsum32(ssv);
    if (l == 0) { atomicAdd(&red[0], sv); atomicAdd(&red[1], ssv); }
  }

  // att @ v_pos (positional, VALU): out[c][p], c group of 4 per half-block
  {
    const int p = tid & 127;
    const int cb = (tid >> 7) * 4;
    float a0 = 0, a1 = 0, a2 = 0, a3 = 0;
    for (int i = 0; i < 128; ++i) {
      const float av = att[p * ATT_STR + i];
      const int d = p - i + 127;
      a0 = fmaf(av, ps[(8 + cb + 0) * 255 + d], a0);
      a1 = fmaf(av, ps[(8 + cb + 1) * 255 + d], a1);
      a2 = fmaf(av, ps[(8 + cb + 2) * 255 + d], a2);
      a3 = fmaf(av, ps[(8 + cb + 3) * 255 + d], a3);
    }
    a0 *= gvp; a1 *= gvp; a2 *= gvp; a3 *= gvp;
    const size_t base = (((size_t)n * 8 + h) * 8 + cb) * 128 + p;
    attvp[base + 0 * 128] = a0;
    attvp[base + 1 * 128] = a1;
    attvp[base + 2 * 128] = a2;
    attvp[base + 3 * 128] = a3;
    float sv = a0 + a1 + a2 + a3;
    float ssv = a0 * a0 + a1 * a1 + a2 * a2 + a3 * a3;
    sv = wsum32(sv); ssv = wsum32(ssv);
    if (l == 0) { atomicAdd(&red[2], sv); atomicAdd(&red[3], ssv); }
  }
  __syncthreads();
  if (tid == 0) {
    atomicAdd(&st[ST_SV + h],      red[0]);
    atomicAdd(&st[ST_SSV + h],     red[1]);
    atomicAdd(&st[ST_SV + 8 + h],  red[2]);
    atomicAdd(&st[ST_SSV + 8 + h], red[3]);
  }
}

// -------------------- pass 7: finalize final BN --------------------
__global__ void k_fin_v(const float* __restrict__ gv2, const float* __restrict__ bv2,
                        float* __restrict__ st) {
  const int t = threadIdx.x;
  if (t < 16) {
    const float cnt = 512.f * 8.f * 128.f;
    const float mean = st[ST_SV + t] / cnt;
    const float var = st[ST_SSV + t] / cnt - mean * mean;
    const float rstd = rsqrtf(var + EPSV);
    const float g = gv2[t];
    st[ST_AV + t] = g * rstd;
    st[ST_BV + t] = bv2[t] - g * mean * rstd;
  }
}

// -------------------- pass 8: final BN + sum + transpose to output --------------------
__global__ __launch_bounds__(256) void k_out(const float* __restrict__ attv,
                                             const float* __restrict__ attvp,
                                             const float* __restrict__ st,
                                             float* __restrict__ out) {
  const int idx = blockIdx.x * 256 + threadIdx.x;
  const int w = idx & 127;
  const int p = (idx >> 7) & 127;
  const int j = (idx >> 14) & 63;
  const int b = idx >> 20;
  const int h = j >> 3, c = j & 7;
  const int n = b * 128 + w;
  const size_t base = (((size_t)n * 8 + h) * 8 + c) * 128 + p;
  out[idx] = st[ST_AV + h] * attv[base] + st[ST_BV + h] +
             st[ST_AV + 8 + h] * attvp[base] + st[ST_BV + 8 + h];
}

extern "C" void kernel_launch(void* const* d_in, const int* in_sizes, int n_in,
                              void* d_out, int out_size, void* d_ws, size_t ws_size,
                              hipStream_t stream) {
  (void)in_sizes; (void)n_in; (void)out_size; (void)ws_size;
  const float* x    = (const float*)d_in[0];
  const float* wq   = (const float*)d_in[1];
  const float* g1   = (const float*)d_in[2];
  const float* b1   = (const float*)d_in[3];
  const float* pr   = (const float*)d_in[4];
  const float* gqr  = (const float*)d_in[5];
  const float* gkr  = (const float*)d_in[6];
  const float* gvp  = (const float*)d_in[7];
  const float* gv   = (const float*)d_in[8];
  const float* gatt = (const float*)d_in[9];
  const float* batt = (const float*)d_in[10];
  const float* gv2  = (const float*)d_in[11];
  const float* bv2  = (const float*)d_in[12];

  char* ws = (char*)d_ws;
  h16*   y     = (h16*)(ws);
  h16*   yn    = (h16*)(ws + 16777216);
  float* attv  = (float*)(ws + 33554432);
  float* attvp = (float*)(ws + 50331648);
  float* st    = (float*)(ws + 67108864);
  float* out   = (float*)d_out;

  k_zero<<<1, 64, 0, stream>>>(st);
  k_gemm_qkv<<<512, 256, 0, stream>>>(x, wq, y);
  k_stats1<<<128, 256, 0, stream>>>(y, g1, b1, st);
  k_bn_transpose<<<512, 256, 0, stream>>>(y, st, yn);
  k_att_stats<<<4096, 256, 0, stream>>>(yn, pr, gqr, gkr, st);
  k_fin_att<<<1, 32, 0, stream>>>(gatt, batt, st);
  k_att_apply<<<4096, 256, SM_BYTES, stream>>>(yn, pr, gqr, gkr, gvp, gv, st, attv, attvp);
  k_fin_v<<<1, 32, 0, stream>>>(gv2, bv2, st);
  k_out<<<16384, 256, 0, stream>>>(attv, attvp, st, out);
}
